// VQVAE_11201274708598
// MI455X (gfx1250) — matrix operand order repaired and checked by reading, not compiled, not run on hardware
//
#include <hip/hip_runtime.h>
#include <hip/hip_bf16.h>
#include <math.h>
#include <stdint.h>

// ---------------------------------------------------------------------------
// VQ-VAE forward for gfx1250 (MI455X).
// All heavy math on matrix cores via V_WMMA_F32_16X16X32_BF16.
// Weights / codebook / activations pre-converted to bf16 (weights zero-padded
// so the GEMM B-side needs no guards); B tiles and VQ tiles staged in LDS with
// GLOBAL_LOAD_ASYNC_TO_LDS_B128 (ASYNCcnt) when the builtin is available.
// ---------------------------------------------------------------------------

typedef __attribute__((ext_vector_type(16))) __bf16 v16bf;
typedef __attribute__((ext_vector_type(8)))  float  v8f;
typedef __attribute__((ext_vector_type(4)))  int    v4i;

union V16 {
    v16bf v;
    unsigned short u[16];
    uint4 q[2];
};

__device__ __forceinline__ unsigned short f2bf(float f) {
    unsigned u = __float_as_uint(f);
    unsigned r = u + 0x7FFFu + ((u >> 16) & 1u);   // round-to-nearest-even
    return (unsigned short)(r >> 16);
}
__device__ __forceinline__ float bf2f(unsigned short h) {
    return __uint_as_float((unsigned)h << 16);
}

#if __has_builtin(__builtin_amdgcn_global_load_async_to_lds_b128)
#define HAVE_ASYNC 1
typedef __attribute__((address_space(1))) v4i gas_v4i;
typedef __attribute__((address_space(3))) v4i las_v4i;
#else
#define HAVE_ASYNC 0
#endif

// Copy 16 bytes global -> LDS. Async (ASYNCcnt) when available.
__device__ __forceinline__ void cp_b128(const void* g, void* l) {
#if HAVE_ASYNC
    __builtin_amdgcn_global_load_async_to_lds_b128(
        (gas_v4i*)(uintptr_t)g,
        (las_v4i*)(unsigned)(uintptr_t)l, 0, 0);
#else
    *(uint4*)l = *(const uint4*)g;
#endif
}
__device__ __forceinline__ void cp_wait() {
#if HAVE_ASYNC
    asm volatile("s_wait_asynccnt 0" ::: "memory");
#endif
}

#define MAX_K 12800      // max Cin*k*k over all layers (512*25)
#define LROW  40         // LDS row stride in shorts (80 B, 16-B aligned)

// ---------------------------------------------------------------------------
// Implicit-GEMM convolution. bf16 activations in (NCHW), bf16 padded weights
// ([Cop][Kp], zero-padded), fp32 out (NCHW). Grid: (M/64, Cop/64); 128 thr.
// `ups`=1: logical input is 4x NN-upsample of a tensor stored at (H/4, W/4).
// ---------------------------------------------------------------------------
__global__ __launch_bounds__(128)
void conv_wmma_k(const unsigned short* __restrict__ in,
                 const unsigned short* __restrict__ wgt,
                 const float* __restrict__ bias, float* __restrict__ out,
                 int Bn, int Cin, int Cout, int H, int W, int k, int pad,
                 int ups, int Kp) {
    __shared__ int ktab[MAX_K];
    __shared__ __align__(16) unsigned short As[64][LROW];
    __shared__ __align__(16) unsigned short Bs[64][LROW];

    const int tid  = threadIdx.x;
    const int lane = tid & 31;
    const int wv   = tid >> 5;
    const int lan  = lane & 15;
    const bool hi  = lane >= 16;

    const int m0 = blockIdx.x * 64;
    const int n0 = blockIdx.y * 64;
    const int HW  = H * W;
    const int kk2 = k * k;
    const int K   = Cin * kk2;

    // decode table: K index -> (cin, kh, kw), once per block
    for (int kk = tid; kk < K; kk += 128) {
        int cin = kk / kk2;
        int r   = kk - cin * kk2;
        int kh  = r / k;
        int kw  = r - kh * k;
        ktab[kk] = (cin << 8) | (kh << 4) | kw;
    }

    // cooperative-loader coordinates (hoisted)
    const int row  = tid >> 1;             // 0..63
    const int col0 = (tid & 1) * 16;
    const int mA   = m0 + row;
    const int bA   = mA / HW;
    const int remA = mA - bA * HW;
    const int ohA  = remA / W;
    const int owA  = remA - ohA * W;

    const int Hs = ups ? (H >> 2) : H;
    const int Ws = ups ? (W >> 2) : W;
    const long long inBase = (long long)bA * Cin * Hs * Ws;
    const unsigned short* wrow = wgt + (long long)(n0 + row) * Kp + col0;

    v8f acc[4] = {};
    const int qa = hi ? 1 : 0;
    const int qb = hi ? 1 : 0;

    __syncthreads();               // ktab ready

    for (int kc = 0; kc < K; kc += 32) {
        // stage B: padded bf16 weights, contiguous, no guards (async b128 x2)
        cp_b128(wrow + kc,     &Bs[row][col0]);
        cp_b128(wrow + kc + 8, &Bs[row][col0 + 8]);
        // stage A: im2col gather of bf16 activations
#pragma unroll
        for (int e = 0; e < 16; ++e) {
            int kk = kc + col0 + e;
            unsigned short av = 0;
            if (kk < K) {
                int tv  = ktab[kk];
                int cin = tv >> 8, kh = (tv >> 4) & 15, kw = tv & 15;
                int ih  = ohA - pad + kh;
                int iw  = owA - pad + kw;
                if (ih >= 0 && ih < H && iw >= 0 && iw < W) {
                    int sih = ups ? (ih >> 2) : ih;
                    int siw = ups ? (iw >> 2) : iw;
                    av = in[inBase + ((long long)cin * Hs + sih) * Ws + siw];
                }
            }
            As[row][col0 + e] = av;
        }
        cp_wait();
        __syncthreads();

        V16 a;
        {
            const uint4* arow = (const uint4*)As[wv * 16 + lan];
            a.q[0] = arow[qa];
            a.q[1] = arow[qa + 2];
        }
#pragma unroll
        for (int ns = 0; ns < 4; ++ns) {
            V16 b;
            const uint4* brow = (const uint4*)Bs[ns * 16 + lan];
            b.q[0] = brow[qb];
            b.q[1] = brow[qb + 2];
            acc[ns] = __builtin_amdgcn_wmma_f32_16x16x32_bf16(
                          false, a.v, false, b.v, (short)0, acc[ns], false, false);
        }
        __syncthreads();
    }

    // epilogue: decode m once per r, reuse across 4 N sub-tiles
#pragma unroll
    for (int r = 0; r < 8; ++r) {
        int m   = m0 + wv * 16 + (hi ? 8 : 0) + r;
        int b2  = m / HW;
        int rem = m - b2 * HW;
        long long obase = (long long)b2 * Cout * HW + rem;
#pragma unroll
        for (int ns = 0; ns < 4; ++ns) {
            int n = n0 + ns * 16 + lan;
            if (n < Cout) out[obase + (long long)n * HW] = acc[ns][r] + bias[n];
        }
    }
}

// ---------------------------------------------------------------------------
// BatchNorm statistics: one block per channel (div-free inner loop).
// ---------------------------------------------------------------------------
__global__ void bn_stats_k(const float* __restrict__ x, float* __restrict__ mean,
                           float* __restrict__ var, int Bn, int C, int HW) {
    int c = blockIdx.x, tid = threadIdx.x;
    float s = 0.f, s2 = 0.f;
    for (int b = 0; b < Bn; ++b) {
        const float* p = x + ((long long)b * C + c) * HW;
        for (int r = tid; r < HW; r += 256) { float v = p[r]; s += v; s2 += v * v; }
    }
    __shared__ float ss[256], ss2[256];
    ss[tid] = s; ss2[tid] = s2;
    __syncthreads();
    for (int st = 128; st > 0; st >>= 1) {
        if (tid < st) { ss[tid] += ss[tid + st]; ss2[tid] += ss2[tid + st]; }
        __syncthreads();
    }
    if (tid == 0) {
        int N = Bn * HW;
        float m = ss[0] / (float)N;
        mean[c] = m;
        var[c]  = ss2[0] / (float)N - m * m;
    }
}

// Normalize + exact GELU; fp32 in place + bf16 copy. Grid (cdiv(HW,256), C, Bn).
__global__ void bn_gelu_k(float* __restrict__ x, unsigned short* __restrict__ xo,
                          const float* __restrict__ mean, const float* __restrict__ var,
                          const float* __restrict__ g, const float* __restrict__ beta,
                          int C, int HW) {
    int r = blockIdx.x * 256 + threadIdx.x;
    if (r >= HW) return;
    int c = blockIdx.y, b = blockIdx.z;
    long long i = ((long long)b * C + c) * HW + r;
    float v = (x[i] - mean[c]) * rsqrtf(var[c] + 1e-5f) * g[c] + beta[c];
    v = 0.5f * v * (1.0f + erff(v * 0.70710678118654752f));
    x[i]  = v;
    xo[i] = f2bf(v);
}

__global__ void maxpool4_k(const unsigned short* __restrict__ in,
                           unsigned short* __restrict__ out, int BC, int H, int W) {
    int Ho = H >> 2, Wo = W >> 2;
    long long total = (long long)BC * Ho * Wo;
    long long i = (long long)blockIdx.x * blockDim.x + threadIdx.x;
    if (i >= total) return;
    int wo = (int)(i % Wo);
    long long t = i / Wo;
    int ho = (int)(t % Ho);
    int bc = (int)(t / Ho);
    const unsigned short* p = in + ((long long)bc * H + ho * 4) * W + wo * 4;
    float m = -3.402823e38f;
    for (int a = 0; a < 4; ++a)
        for (int b = 0; b < 4; ++b) { float v = bf2f(p[a * W + b]); m = v > m ? v : m; }
    out[i] = f2bf(m);
}

// ---------------------------------------------------------------------------
// Conversions
// ---------------------------------------------------------------------------
__global__ void f32_to_bf16_k(const float* __restrict__ in,
                              unsigned short* __restrict__ out, long long n) {
    long long i = (long long)blockIdx.x * blockDim.x + threadIdx.x;
    if (i < n) out[i] = f2bf(in[i]);
}

// Zero-padded bf16 weights: [Cop][Kp] from [Cout][K].
__global__ void wpad_bf16_k(const float* __restrict__ w, unsigned short* __restrict__ out,
                            int Cout, int K, int Cop, int Kp) {
    long long i = (long long)blockIdx.x * blockDim.x + threadIdx.x;
    long long total = (long long)Cop * Kp;
    if (i >= total) return;
    int co = (int)(i / Kp);
    int kk = (int)(i - (long long)co * Kp);
    out[i] = (co < Cout && kk < K) ? f2bf(w[(long long)co * K + kk]) : 0;
}

// z (4,1024,8,8) NCHW fp32 -> zflat bf16 [256][1024] row-major.
__global__ void zflat_bf16_k(const float* __restrict__ z, unsigned short* __restrict__ zf) {
    int row = blockIdx.x;                               // 256
    int b = row >> 6, h = (row >> 3) & 7, w = row & 7;
    int base = b * 65536 + h * 8 + w;
    for (int c = threadIdx.x; c < 1024; c += 256)
        zf[row * 1024 + c] = f2bf(z[base + c * 64]);
}

// ---------------------------------------------------------------------------
// VQ distance GEMM M=256 x N=65536 x K=1024 with fused argmin.
// A = zflat bf16 (row-major), B = bf16 codebook; both async-staged.
// Grid (4, 1024); 128 threads.
// ---------------------------------------------------------------------------
__global__ __launch_bounds__(128)
void vq_argmin_k(const unsigned short* __restrict__ zf,
                 const unsigned short* __restrict__ cb,
                 const float* __restrict__ rn, const float* __restrict__ cn,
                 unsigned long long* __restrict__ slots) {
    __shared__ __align__(16) unsigned short As[64][LROW];
    __shared__ __align__(16) unsigned short Bs[64][LROW];

    const int tid  = threadIdx.x;
    const int lane = tid & 31;
    const int wv   = tid >> 5;
    const int lan  = lane & 15;
    const bool hi  = lane >= 16;

    const int m0 = blockIdx.x * 64;
    const int n0 = blockIdx.y * 64;

    const int row  = tid >> 1;
    const int col0 = (tid & 1) * 16;
    const unsigned short* arow_g = zf + (long long)(m0 + row) * 1024 + col0;
    const unsigned short* brow_g = cb + (long long)(n0 + row) * 1024 + col0;

    v8f acc[4] = {};
    const int qa = hi ? 1 : 0;
    const int qb = hi ? 1 : 0;

    for (int kc = 0; kc < 1024; kc += 32) {
        cp_b128(arow_g + kc,     &As[row][col0]);
        cp_b128(arow_g + kc + 8, &As[row][col0 + 8]);
        cp_b128(brow_g + kc,     &Bs[row][col0]);
        cp_b128(brow_g + kc + 8, &Bs[row][col0 + 8]);
        cp_wait();
        __syncthreads();

        V16 a;
        {
            const uint4* arow = (const uint4*)As[wv * 16 + lan];
            a.q[0] = arow[qa];
            a.q[1] = arow[qa + 2];
        }
#pragma unroll
        for (int ns = 0; ns < 4; ++ns) {
            V16 b;
            const uint4* brow = (const uint4*)Bs[ns * 16 + lan];
            b.q[0] = brow[qb];
            b.q[1] = brow[qb + 2];
            acc[ns] = __builtin_amdgcn_wmma_f32_16x16x32_bf16(
                          false, a.v, false, b.v, (short)0, acc[ns], false, false);
        }
        __syncthreads();
    }

#pragma unroll
    for (int ns = 0; ns < 4; ++ns) {
#pragma unroll
        for (int r = 0; r < 8; ++r) {
            int m = m0 + wv * 16 + (hi ? 8 : 0) + r;
            int n = n0 + ns * 16 + lan;
            float d = rn[m] + cn[n] - 2.0f * acc[ns][r];
            unsigned u = __float_as_uint(d);
            u = (u & 0x80000000u) ? ~u : (u | 0x80000000u);  // order-preserving map
            unsigned long long key = ((unsigned long long)u << 32) | (unsigned)n;
            atomicMin(&slots[m], key);
        }
    }
}

__global__ void codenorm_k(const float* __restrict__ cb, float* __restrict__ cn) {
    int code = blockIdx.x * blockDim.x + threadIdx.x;   // 65536
    const float* p = cb + (long long)code * 1024;
    float s = 0.f;
    for (int c = 0; c < 1024; ++c) { float v = p[c]; s += v * v; }
    cn[code] = s;
}

__global__ void rownorm_k(const float* __restrict__ z, float* __restrict__ rn) {
    int row = threadIdx.x;                              // 256
    int b = row >> 6, h = (row >> 3) & 7, w = row & 7;
    int base = b * 65536 + h * 8 + w;
    float s = 0.f;
    for (int c = 0; c < 1024; ++c) { float v = z[base + c * 64]; s += v * v; }
    rn[row] = s;
}

__global__ void vq_init_k(unsigned long long* __restrict__ slots, float* __restrict__ acc) {
    int i = threadIdx.x;                                // 256
    slots[i] = 0xFFFFFFFFFFFFFFFFull;
    if (i == 0) acc[0] = 0.f;
}

__global__ void vq_gather_k(const float* __restrict__ z, const float* __restrict__ cb,
                            const unsigned long long* __restrict__ slots,
                            unsigned short* __restrict__ qbf, float* __restrict__ lossAcc,
                            float* __restrict__ outIdx) {
    int row = blockIdx.x, tid = threadIdx.x;            // 256 x 256
    int idx = (int)(slots[row] & 0xFFFFFFFFull);
    int b = row >> 6, h = (row >> 3) & 7, w = row & 7;
    float s = 0.f;
    for (int c = tid; c < 1024; c += 256) {
        float qa = cb[(long long)idx * 1024 + c];
        int za = b * 65536 + c * 64 + h * 8 + w;
        float d = qa - z[za];
        qbf[za] = f2bf(qa);
        s += d * d;
    }
    __shared__ float ss[256];
    ss[tid] = s;
    __syncthreads();
    for (int st = 128; st > 0; st >>= 1) {
        if (tid < st) ss[tid] += ss[tid + st];
        __syncthreads();
    }
    if (tid == 0) {
        atomicAdd(lossAcc, ss[0]);
        outIdx[row] = (float)idx;
    }
}

__global__ void finalize_loss_k(const float* __restrict__ acc, float* __restrict__ out) {
    out[0] = 1.25f * acc[0] / 262144.0f;   // q_loss + 0.25*e_loss (forward-equal)
}

__global__ void sigmoid_k(const float* __restrict__ in, float* __restrict__ out, int n) {
    int i = blockIdx.x * blockDim.x + threadIdx.x;
    if (i < n) out[i] = 1.0f / (1.0f + expf(-in[i]));
}

// ---------------------------------------------------------------------------
// Host orchestration
// ---------------------------------------------------------------------------
struct Spec { int cin, cout, k; };

static inline int cdiv(long long a, long long b) { return (int)((a + b - 1) / b); }

extern "C" void kernel_launch(void* const* d_in, const int* in_sizes, int n_in,
                              void* d_out, int out_size, void* d_ws, size_t ws_size,
                              hipStream_t stream) {
    (void)in_sizes; (void)n_in; (void)out_size; (void)ws_size;

    // Inputs in setup_inputs() dict insertion order:
    // x, codebook, enc[0..5]{w,b,g,beta}, dec[0..5]{w,b,g,beta}
    const float* x        = (const float*)d_in[0];
    const float* codebook = (const float*)d_in[1];
    const float* lp[12][4];
    for (int i = 0; i < 12; ++i)
        for (int j = 0; j < 4; ++j) lp[i][j] = (const float*)d_in[2 + i * 4 + j];

    const Spec spec[12] = {
        {1,64,7},{64,512,5},{512,1024,3},{1024,1024,1},{1024,1024,3},{1024,1024,3},
        {1024,1024,3},{1024,1024,3},{1024,1024,1},{1024,512,3},{512,64,5},{64,1,5}};
    const int Bn = 4;

    // -------- workspace carve-up (float units) --------
    float* ws      = (float*)d_ws;
    float* convF32 = ws;                                   // 67,108,864 f32
    unsigned short* actA = (unsigned short*)(convF32 + 67108864LL);  // 67,108,864 bf16
    unsigned short* actB = actA + 67108864LL;              // 67,108,864 bf16
    unsigned short* cbbf = actB + 67108864LL;              // 67,108,864 bf16
    unsigned short* wbf  = cbbf + 67108864LL;              // padded bf16 weights
    // per-layer padded weight offsets
    unsigned short* wptr[12];
    long long woff = 0;
    int Kp[12], Cop[12];
    for (int i = 0; i < 12; ++i) {
        int K = spec[i].cin * spec[i].k * spec[i].k;
        Kp[i]  = ((K + 31) / 32) * 32;
        Cop[i] = ((spec[i].cout + 63) / 64) * 64;
        wptr[i] = wbf + woff;
        woff += (long long)Kp[i] * Cop[i];
    }
    float* tail  = (float*)(wbf + ((woff + 1) & ~1LL));
    float* zbuf  = tail;                                   // 262,144 f32
    unsigned short* zflat = (unsigned short*)(zbuf + 262144);  // 262,144 bf16
    unsigned short* qbf   = zflat + 262144;                // 262,144 bf16
    float* cnorm = (float*)(qbf + 262144);                 // 65,536
    float* rnorm = cnorm + 65536;                          // 256
    unsigned long long* slots = (unsigned long long*)(rnorm + 256);  // 256 u64
    float* meanb = (float*)(slots + 256);                  // 1024
    float* varb  = meanb + 1024;                           // 1024
    float* lossA = varb + 1024;                            // 1

    float* recon_out = (float*)d_out;                      // 1,048,576
    float* loss_out  = recon_out + 1048576;
    float* idx_out   = loss_out + 1;

    // -------- one-time conversions --------
    for (int i = 0; i < 12; ++i) {
        int K = spec[i].cin * spec[i].k * spec[i].k;
        long long total = (long long)Cop[i] * Kp[i];
        wpad_bf16_k<<<cdiv(total, 256), 256, 0, stream>>>(lp[i][0], wptr[i],
                                                          spec[i].cout, K, Cop[i], Kp[i]);
    }
    f32_to_bf16_k<<<cdiv(67108864LL, 256), 256, 0, stream>>>(codebook, cbbf, 67108864LL);
    f32_to_bf16_k<<<cdiv(1048576LL, 256), 256, 0, stream>>>(x, actA, 1048576LL);

    unsigned short* act[2] = {actA, actB};

    // ---------------- Encoder ----------------
    const unsigned short* curb = actA;
    int H = 512, C = 1, pp = 1;
    for (int i = 0; i < 6; ++i) {
        int Co = spec[i].cout, k = spec[i].k;
        long long M = (long long)Bn * H * H;
        dim3 grid(cdiv(M, 64), Cop[i] / 64);
        conv_wmma_k<<<grid, 128, 0, stream>>>(curb, wptr[i], lp[i][1], convF32,
                                              Bn, C, Co, H, H, k, k / 2, 0, Kp[i]);
        bn_stats_k<<<Co, 256, 0, stream>>>(convF32, meanb, varb, Bn, Co, H * H);
        dim3 g2(cdiv(H * H, 256), Co, Bn);
        bn_gelu_k<<<g2, 256, 0, stream>>>(convF32, act[pp], meanb, varb,
                                          lp[i][2], lp[i][3], Co, H * H);
        if (i < 3) {
            long long total = (long long)Bn * Co * (H / 4) * (H / 4);
            maxpool4_k<<<cdiv(total, 256), 256, 0, stream>>>(act[pp], act[1 - pp],
                                                             Bn * Co, H, H);
            H /= 4;
            curb = act[1 - pp];
        } else {
            curb = act[pp];
            pp = 1 - pp;
        }
        C = Co;
    }
    // convF32 currently holds fp32 z (post-GELU, layer 5); keep a copy
    (void)hipMemcpyAsync(zbuf, convF32, 262144 * sizeof(float),
                         hipMemcpyDeviceToDevice, stream);

    // ---------------- Vector quantization ----------------
    codenorm_k<<<256, 256, 0, stream>>>(codebook, cnorm);
    rownorm_k<<<1, 256, 0, stream>>>(zbuf, rnorm);
    zflat_bf16_k<<<256, 256, 0, stream>>>(zbuf, zflat);
    vq_init_k<<<1, 256, 0, stream>>>(slots, lossA);
    {
        dim3 grid(4, 1024);
        vq_argmin_k<<<grid, 128, 0, stream>>>(zflat, cbbf, rnorm, cnorm, slots);
    }
    vq_gather_k<<<256, 256, 0, stream>>>(zbuf, codebook, slots, qbf, lossA, idx_out);
    finalize_loss_k<<<1, 1, 0, stream>>>(lossA, loss_out);

    // ---------------- Decoder ----------------
    curb = qbf; H = 8; C = 1024; pp = 0;
    for (int i = 0; i < 6; ++i) {
        int li = 6 + i;
        int Co = spec[li].cout, k = spec[li].k;
        int ups = (i >= 3) ? 1 : 0;
        if (ups) H *= 4;
        long long M = (long long)Bn * H * H;
        dim3 grid(cdiv(M, 64), Cop[li] / 64);
        conv_wmma_k<<<grid, 128, 0, stream>>>(curb, wptr[li], lp[li][1], convF32,
                                              Bn, C, Co, H, H, k, k / 2, ups, Kp[li]);
        if (i < 5) {
            bn_stats_k<<<Co, 256, 0, stream>>>(convF32, meanb, varb, Bn, Co, H * H);
            dim3 g2(cdiv(H * H, 256), Co, Bn);
            bn_gelu_k<<<g2, 256, 0, stream>>>(convF32, act[pp], meanb, varb,
                                              lp[li][2], lp[li][3], Co, H * H);
            curb = act[pp];
            pp = 1 - pp;
        } else {
            int n = Bn * Co * H * H;   // 4*1*512*512
            sigmoid_k<<<cdiv(n, 256), 256, 0, stream>>>(convF32, recon_out, n);
        }
        C = Co;
    }
}
